// AltAttention_1726576856257
// MI455X (gfx1250) — matrix operand order repaired and checked by reading, not compiled, not run on hardware
//
#include <hip/hip_runtime.h>
#include <hip/hip_fp16.h>

// MI455X / gfx1250, wave32. Flash-attention with cosine-sim + alibi + mask.
// Matmuls: v_wmma_f32_16x16x32_f16 (f16 in, f32 accum), software-pipelined.
// Softmax kept in log2 domain (scale*log2e folded into q at QKV epilogue);
// alibi*log2e + mask-bias fused into the WMMA C-accumulator init, so the
// per-entry hot path is just max / sub / exp2 / sum.

typedef __attribute__((ext_vector_type(16))) _Float16 v16h;
typedef __attribute__((ext_vector_type(8)))  _Float16 v8h;
typedef __attribute__((ext_vector_type(4)))  _Float16 v4h;
typedef __attribute__((ext_vector_type(8)))  float    v8f;
typedef __attribute__((ext_vector_type(4)))  float    v4f;

#define Bb 2
#define NSEQ 2048
#define CC 1024
#define HH 16
#define DD 64
#define LOG_MAX 4.605170185988091f   // log(1/0.01)
#define L2E 1.4426950408889634f      // log2(e)

// ---------------------------------------------------------------------------
// WMMA fragment loaders (CDNA5 layouts, CDNA5 ISA §7.12.2)
// ---------------------------------------------------------------------------
__device__ __forceinline__ v16h load_frag_a(const _Float16* __restrict__ src,
                                            int ld, int m0, int k0, int lane) {
  int row = m0 + (lane & 15);
  int kb  = (lane >> 4) << 3;                  // 0 or 8
  const _Float16* p = src + (size_t)row * ld + k0 + kb;
  v8h lo = *(const v8h*)(p);                   // K = kb .. kb+7
  v8h hi = *(const v8h*)(p + 16);              // K = kb+16 .. kb+23
  v16h a;
#pragma unroll
  for (int i = 0; i < 8; ++i) { a[i] = lo[i]; a[8 + i] = hi[i]; }
  return a;
}

// B[k][n] = src[n][k], src row-major (stride ld): cols n0..n0+15, K k0..k0+31.
__device__ __forceinline__ v16h load_frag_b(const _Float16* __restrict__ src,
                                            int ld, int n0, int k0, int lane) {
  int col = n0 + (lane & 15);
  int kb  = (lane >> 4) << 3;                  // 0 or 8
  const _Float16* p = src + (size_t)col * ld + k0 + kb;
  v8h lo = *(const v8h*)(p);
  v8h hi = *(const v8h*)(p + 16);
  v16h b;
#pragma unroll
  for (int i = 0; i < 8; ++i) { b[i] = lo[i]; b[8 + i] = hi[i]; }
  return b;
}

#define WMMA(A, B, C)                                                         \
  __builtin_amdgcn_wmma_f32_16x16x32_f16(false, (A), false, (B), (short)0,    \
                                         (C), false, false)

// ---------------------------------------------------------------------------
// f32 -> f16 convert (vector 4-wide)
// ---------------------------------------------------------------------------
__global__ __launch_bounds__(256) void cvt_f32_f16(const float* __restrict__ s,
                                                   _Float16* __restrict__ d,
                                                   int n) {
  int i = (blockIdx.x * blockDim.x + threadIdx.x) * 4;
  if (i < n) {
    v4f v = *(const v4f*)(s + i);
    v4h o;
#pragma unroll
    for (int e = 0; e < 4; ++e) o[e] = (_Float16)v[e];
    *(v4h*)(d + i) = o;
  }
}

// key-padding mask -> additive bias (0 or -1e30), per (b, k)
__global__ __launch_bounds__(256)
void mask_bias(const unsigned char* __restrict__ mk, float* __restrict__ mb,
               int n) {
  int i = blockIdx.x * blockDim.x + threadIdx.x;
  if (i < n) mb[i] = mk[i] ? -1e30f : 0.0f;
}

// ---------------------------------------------------------------------------
// QKV GEMM: 32 rows x 64 cols per wave (one (q|k|v, head) D-slice),
// double-buffered k-loop.  Cosine normalization fused; q additionally scaled
// by exp(min(logit_scale,LOG_MAX))*log2e so attention runs in log2 domain.
// q,k stored [B,H,N,D]; v stored TRANSPOSED [B,H,D,N].
// ---------------------------------------------------------------------------
__global__ __launch_bounds__(256)
void qkv_gemm(const _Float16* __restrict__ xh, const _Float16* __restrict__ wh,
              const float* __restrict__ lscale, _Float16* __restrict__ qn,
              _Float16* __restrict__ kn, _Float16* __restrict__ vt) {
  const int lane = threadIdx.x & 31;
  const int wid  = blockIdx.x * (blockDim.x >> 5) + (threadIdx.x >> 5);
  const int NB = (3 * CC) / 64;                // 48 n-blocks
  const int nb = wid % NB, mb = wid / NB;
  const int m0 = mb * 32, n0 = nb * 64;

  v8f acc[2][4] = {};
  v16h a0 = load_frag_a(xh, CC, m0, 0, lane);
  v16h a1 = load_frag_a(xh, CC, m0 + 16, 0, lane);
  v16h bf[4];
#pragma unroll
  for (int t = 0; t < 4; ++t) bf[t] = load_frag_b(wh, CC, n0 + 16 * t, 0, lane);

#pragma unroll 2
  for (int k0 = 0; k0 < CC; k0 += 32) {
    const int k1 = (k0 + 32 < CC) ? k0 + 32 : 0;   // wrap: dead loads on last
    v16h na0 = load_frag_a(xh, CC, m0, k1, lane);
    v16h na1 = load_frag_a(xh, CC, m0 + 16, k1, lane);
    v16h nb_[4];
#pragma unroll
    for (int t = 0; t < 4; ++t)
      nb_[t] = load_frag_b(wh, CC, n0 + 16 * t, k1, lane);
#pragma unroll
    for (int t = 0; t < 4; ++t) acc[0][t] = WMMA(a0, bf[t], acc[0][t]);
#pragma unroll
    for (int t = 0; t < 4; ++t) acc[1][t] = WMMA(a1, bf[t], acc[1][t]);
    a0 = na0; a1 = na1;
#pragma unroll
    for (int t = 0; t < 4; ++t) bf[t] = nb_[t];
  }

  const int which = n0 >> 10;                  // 0=q 1=k 2=v
  const int h     = (n0 & (CC - 1)) >> 6;
  const int col   = lane & 15;
  const int rbase = (lane >> 4) << 3;
  const int b     = m0 >> 11;
  const int ns0   = m0 & (NSEQ - 1);

  if (which < 2) {                             // cosine normalize over D
    _Float16* dst = (which == 0) ? qn : kn;
    // fold per-head logit scale (and log2e) into q
    const float post =
        (which == 0) ? __expf(fminf(lscale[h], LOG_MAX)) * L2E : 1.0f;
    const size_t base = ((size_t)b * HH + h) * NSEQ;
#pragma unroll
    for (int ms = 0; ms < 2; ++ms) {
#pragma unroll
      for (int j = 0; j < 8; ++j) {
        float ss = 0.f;
#pragma unroll
        for (int t = 0; t < 4; ++t) ss += acc[ms][t][j] * acc[ms][t][j];
        ss += __shfl_xor(ss, 1, 32);
        ss += __shfl_xor(ss, 2, 32);
        ss += __shfl_xor(ss, 4, 32);
        ss += __shfl_xor(ss, 8, 32);
        float inv = rsqrtf(ss + 1e-20f) * post;
        _Float16* o = dst + (base + ns0 + 16 * ms + rbase + j) * DD;
#pragma unroll
        for (int t = 0; t < 4; ++t)
          o[16 * t + col] = (_Float16)(acc[ms][t][j] * inv);
      }
    }
  } else {                                     // v: transposed, vector stores
#pragma unroll
    for (int ms = 0; ms < 2; ++ms) {
#pragma unroll
      for (int t = 0; t < 4; ++t) {
        v8h pk;
#pragma unroll
        for (int j = 0; j < 8; ++j) pk[j] = (_Float16)acc[ms][t][j];
        _Float16* o = vt +
            ((size_t)(b * HH + h) * DD + 16 * t + col) * NSEQ +
            ns0 + 16 * ms + rbase;
        *(v8h*)o = pk;                         // 8 consecutive n: one b128
      }
    }
  }
}

// ---------------------------------------------------------------------------
// Flash attention. One wave per 16-row q-tile; S^T = K·Q^T so softmaxed P
// registers ARE the A-fragment of P·V.  Alibi (non-temporal, double-buffered
// HBM stream) lands directly in the WMMA C accumulator; mask bias fused with
// 4 pk_fma.  O-rescale skipped via ballot when the running max didn't grow.
// ---------------------------------------------------------------------------
__global__ __launch_bounds__(256)
void attn_kernel(const _Float16* __restrict__ qn, const _Float16* __restrict__ kn,
                 const _Float16* __restrict__ vt, const float* __restrict__ alibi,
                 const float* __restrict__ mbias, _Float16* __restrict__ oh) {
  const int lane = threadIdx.x & 31;
  const int wid  = blockIdx.x * 8 + (threadIdx.x >> 5);
  const int qt   = wid & 127;                  // N/16 q-tiles
  const int bh   = wid >> 7;                   // 0..31
  const int b    = bh >> 4;
  const int q0   = qt * 16;

  const _Float16* Q  = qn + (size_t)bh * NSEQ * DD;
  const _Float16* K  = kn + (size_t)bh * NSEQ * DD;
  const _Float16* VT = vt + (size_t)bh * DD * NSEQ;
  const float*    AL = alibi + ((size_t)bh * NSEQ + q0) * NSEQ;
  const float*    MB = mbias + b * NSEQ;

  const v16h qf0 = load_frag_b(Q, DD, q0, 0, lane);   // Q^T B-frag, d 0..31
  const v16h qf1 = load_frag_b(Q, DD, q0, 32, lane);  // d 32..63

  const int qcol = lane & 15;
  const int kb   = (lane >> 4) << 3;           // 0 or 8  (A-frag half split)
  const int Kb   = (lane >> 4) << 4;           // 0 or 16 (unused)
  const int rsel = (lane & 16) >> 1;           // row base for O broadcast

  float m_run = -1e30f, l_run = 0.f;
  v8f O[4] = {};

  // ---- pre-load chunk 0: K fragments + alibi rows (non-temporal) ----
  const float* alp = AL + (size_t)qcol * NSEQ + kb;
  v16h kf0 = load_frag_a(K, DD, 0, 0, lane);
  v16h kf1 = load_frag_a(K, DD, 0, 32, lane);
  v16h kf2 = load_frag_a(K, DD, 16, 0, lane);
  v16h kf3 = load_frag_a(K, DD, 16, 32, lane);
  v8f alA = __builtin_nontemporal_load((const v8f*)(alp));       // k kb..kb+7
  v8f alB = __builtin_nontemporal_load((const v8f*)(alp + 16));  // +16

  for (int kc = 0; kc < NSEQ; kc += 32) {
    // ---- current-chunk V^T B-fragments (L2-resident, issued early) ----
    v16h vf[4];
#pragma unroll
    for (int t = 0; t < 4; ++t) {
      const _Float16* p = VT + (size_t)(16 * t + qcol) * NSEQ + kc + kb;
      v8h lo = *(const v8h*)p;
      v8h hi = *(const v8h*)(p + 16);
#pragma unroll
      for (int i = 0; i < 8; ++i) { vf[t][i] = lo[i]; vf[t][8 + i] = hi[i]; }
    }
    // mask bias for this chunk (tiny, L2-resident)
    v8f mb0 = *(const v8f*)(MB + kc + kb);
    v8f mb1 = *(const v8f*)(MB + kc + 16 + kb);

    // ---- prefetch next chunk (wrap on last iteration: dead loads) ----
    const int kc2 = (kc + 32 < NSEQ) ? kc + 32 : 0;
    v16h nk0 = load_frag_a(K, DD, kc2, 0, lane);
    v16h nk1 = load_frag_a(K, DD, kc2, 32, lane);
    v16h nk2 = load_frag_a(K, DD, kc2 + 16, 0, lane);
    v16h nk3 = load_frag_a(K, DD, kc2 + 16, 32, lane);
    v8f nalA = __builtin_nontemporal_load((const v8f*)(alp + kc2));
    v8f nalB = __builtin_nontemporal_load((const v8f*)(alp + kc2 + 16));

    // ---- C-accumulator init = alibi*log2e + mask_bias (4+4 pk_fma) ----
    v8f C0, C1;
#pragma unroll
    for (int i = 0; i < 8; ++i) {
      C0[i] = fmaf(alA[i], L2E, mb0[i]);
      C1[i] = fmaf(alB[i], L2E, mb1[i]);
    }

    // ---- S^T(log2) = K·Q^T + C : two 16x16 tiles over k ----
    v8f T0 = WMMA(kf0, qf0, C0);
    T0 = WMMA(kf1, qf1, T0);
    v8f T1 = WMMA(kf2, qf0, C1);
    T1 = WMMA(kf3, qf1, T1);

    // ---- online softmax over k (fixed q = lane): local + xor-16 ----
    float mx = fmaxf(T0[0], T0[1]);
#pragma unroll
    for (int i = 2; i < 8; ++i) mx = fmaxf(mx, T0[i]);
#pragma unroll
    for (int i = 0; i < 8; ++i) mx = fmaxf(mx, T1[i]);
    mx = fmaxf(mx, __shfl_xor(mx, 16, 32));
    const float m_new = fmaxf(m_run, mx);
    const float alpha = exp2f(m_run - m_new);
    const bool  grew  = m_new > m_run;
    m_run = m_new;

    float psum = 0.f;
    v16h pf;                                   // directly the A-fragment of P
#pragma unroll
    for (int i = 0; i < 8; ++i) {
      float p = exp2f(T0[i] - m_new);
      psum += p;
      pf[i] = (_Float16)p;
    }
#pragma unroll
    for (int i = 0; i < 8; ++i) {
      float p = exp2f(T1[i] - m_new);
      psum += p;
      pf[8 + i] = (_Float16)p;
    }
    psum += __shfl_xor(psum, 16, 32);
    l_run = l_run * alpha + psum;

    // ---- rescale O only when some row's max grew (rare after warmup) ----
    if (__ballot(grew)) {
#pragma unroll
      for (int j = 0; j < 8; ++j) {
        float aj = __shfl(alpha, rsel + j, 32);
#pragma unroll
        for (int t = 0; t < 4; ++t) O[t][j] *= aj;
      }
    }

    // ---- O += P · V ----
#pragma unroll
    for (int t = 0; t < 4; ++t) O[t] = WMMA(pf, vf[t], O[t]);

    // ---- rotate double buffers ----
    kf0 = nk0; kf1 = nk1; kf2 = nk2; kf3 = nk3;
    alA = nalA; alB = nalB;
  }

  // ---- finalize: divide by l, store f16 to [B][N][H*D] for proj GEMM ----
  const float linv = 1.f / fmaxf(l_run, 1e-30f);
  const int rbase = (lane >> 4) << 3;
  const int h = bh & (HH - 1);
#pragma unroll
  for (int j = 0; j < 8; ++j) {
    float lj = __shfl(linv, rsel + j, 32);
    _Float16* op = oh + ((size_t)(b * NSEQ + q0 + rbase + j)) * CC + h * DD;
#pragma unroll
    for (int t = 0; t < 4; ++t) op[16 * t + qcol] = (_Float16)(O[t][j] * lj);
  }
}

// ---------------------------------------------------------------------------
// Output projection: [4096,1024]f16 x [1024,1024]f16^T + bias -> f32 out.
// 32x64 per wave, pipelined; non-temporal f32 stores (write-once output).
// ---------------------------------------------------------------------------
__global__ __launch_bounds__(256)
void proj_gemm(const _Float16* __restrict__ oh, const _Float16* __restrict__ pwh,
               const float* __restrict__ pb, float* __restrict__ out) {
  const int lane = threadIdx.x & 31;
  const int wid  = blockIdx.x * (blockDim.x >> 5) + (threadIdx.x >> 5);
  const int NB = CC / 64;                      // 16
  const int nb = wid % NB, mb = wid / NB;
  const int m0 = mb * 32, n0 = nb * 64;

  v8f acc[2][4] = {};
  v16h a0 = load_frag_a(oh, CC, m0, 0, lane);
  v16h a1 = load_frag_a(oh, CC, m0 + 16, 0, lane);
  v16h bf[4];
#pragma unroll
  for (int t = 0; t < 4; ++t)
    bf[t] = load_frag_b(pwh, CC, n0 + 16 * t, 0, lane);

#pragma unroll 2
  for (int k0 = 0; k0 < CC; k0 += 32) {
    const int k1 = (k0 + 32 < CC) ? k0 + 32 : 0;
    v16h na0 = load_frag_a(oh, CC, m0, k1, lane);
    v16h na1 = load_frag_a(oh, CC, m0 + 16, k1, lane);
    v16h nb_[4];
#pragma unroll
    for (int t = 0; t < 4; ++t)
      nb_[t] = load_frag_b(pwh, CC, n0 + 16 * t, k1, lane);
#pragma unroll
    for (int t = 0; t < 4; ++t) acc[0][t] = WMMA(a0, bf[t], acc[0][t]);
#pragma unroll
    for (int t = 0; t < 4; ++t) acc[1][t] = WMMA(a1, bf[t], acc[1][t]);
    a0 = na0; a1 = na1;
#pragma unroll
    for (int t = 0; t < 4; ++t) bf[t] = nb_[t];
  }

  const int col = lane & 15;
  const int rbase = (lane >> 4) << 3;
  float bias[4];
#pragma unroll
  for (int t = 0; t < 4; ++t) bias[t] = pb[n0 + 16 * t + col];
#pragma unroll
  for (int ms = 0; ms < 2; ++ms) {
#pragma unroll
    for (int j = 0; j < 8; ++j) {
      float* o = out + (size_t)(m0 + 16 * ms + rbase + j) * CC + n0;
#pragma unroll
      for (int t = 0; t < 4; ++t)
        __builtin_nontemporal_store(acc[ms][t][j] + bias[t], o + 16 * t + col);
    }
  }
}

// ---------------------------------------------------------------------------
extern "C" void kernel_launch(void* const* d_in, const int* in_sizes, int n_in,
                              void* d_out, int out_size, void* d_ws,
                              size_t ws_size, hipStream_t stream) {
  const float* x   = (const float*)d_in[0];                 // [2,2048,1024]
  const unsigned char* pmask = (const unsigned char*)d_in[1]; // [2,2048] bool
  const float* alibi = (const float*)d_in[2];               // [2,16,2048,2048]
  const float* qkv_w = (const float*)d_in[3];               // [3072,1024]
  const float* proj_w = (const float*)d_in[4];              // [1024,1024]
  const float* proj_b = (const float*)d_in[5];              // [1024]
  const float* lscale = (const float*)d_in[6];              // [16]
  float* out = (float*)d_out;

  const size_t M = (size_t)Bb * NSEQ;                       // 4096
  char* w = (char*)d_ws;
  auto take = [&](size_t bytes) {
    char* p = w;
    w += (bytes + 255) & ~(size_t)255;
    return p;
  };
  _Float16* xh  = (_Float16*)take(M * CC * 2);              // 8 MB
  _Float16* wh  = (_Float16*)take((size_t)3 * CC * CC * 2); // 6 MB
  _Float16* pwh = (_Float16*)take((size_t)CC * CC * 2);     // 2 MB
  _Float16* qn  = (_Float16*)take(M * CC * 2);              // 8 MB
  _Float16* kn  = (_Float16*)take(M * CC * 2);              // 8 MB
  _Float16* vt  = (_Float16*)take(M * CC * 2);              // 8 MB (B,H,D,N)
  _Float16* oh  = (_Float16*)take(M * CC * 2);              // 8 MB
  float*    mb  = (float*)take(M * 4);                      // 16 KB

  // f32 -> f16 converts + mask bias
  cvt_f32_f16<<<(int)(M * CC / 4 / 256), 256, 0, stream>>>(x, xh,
                                                           (int)(M * CC));
  cvt_f32_f16<<<(3 * CC * CC / 4) / 256, 256, 0, stream>>>(qkv_w, wh,
                                                           3 * CC * CC);
  cvt_f32_f16<<<(CC * CC / 4) / 256, 256, 0, stream>>>(proj_w, pwh, CC * CC);
  mask_bias<<<(int)(M / 256), 256, 0, stream>>>(pmask, mb, (int)M);

  // QKV projection + fused normalize/scale: 128 m-blocks * 48 n-blocks waves
  qkv_gemm<<<(128 * 48) / 8, 256, 0, stream>>>(xh, wh, lscale, qn, kn, vt);

  // Flash attention: 2*16*128 = 4096 q-tile waves
  attn_kernel<<<4096 / 8, 256, 0, stream>>>(qn, kn, vt, alibi, mb, oh);

  // Output projection: 128 m-blocks * 16 n-blocks waves
  proj_gemm<<<(128 * 16) / 8, 256, 0, stream>>>(oh, pwh, proj_b, out);
}
